// Mambadelta_76708115907016
// MI455X (gfx1250) — matrix operand order repaired and checked by reading, not compiled, not run on hardware
//
#include <hip/hip_runtime.h>
#include <cstdint>
#include <cstddef>

typedef __attribute__((ext_vector_type(16))) __bf16 v16bf;
typedef __attribute__((ext_vector_type(8)))  __bf16 v8bf;
typedef __attribute__((ext_vector_type(8)))  float   v8f;

#define BB 2
#define LL 512
#define BL (BB * LL)          // 1024 : M dimension of every GEMM

__device__ __forceinline__ float sigmoidf_(float x) { return 1.0f / (1.0f + expf(-x)); }
__device__ __forceinline__ float softplusf_(float x) { return x > 20.0f ? x : log1pf(expf(x)); }

// ---------------------------------------------------------------------------
// Generic C = A(MxK,bf16) * W(NxK,bf16)^T  (+ bias_scale*bias[col], + act)
// wave tile 32x32 (4 wmma accs), block = 8 waves (2M x 4N) -> 64x128 tile.
// A fragment (16-bit A 16x32, ISA 7.12.2): lane m      : K = {0..7, 16..23}
//                                          lane m+16   : K = {8..15, 24..31}
// B fragment (16-bit B 32x16):             lane n      : K = {0..7, 16..23}
//                                          lane n+16   : K = {8..15, 24..31}
// ---------------------------------------------------------------------------
__device__ __forceinline__ v16bf load_afrag(const __bf16* p) {
  v8bf lo = *reinterpret_cast<const v8bf*>(p);        // 8 bf16 = 16 bytes
  v8bf hi = *reinterpret_cast<const v8bf*>(p + 16);   // +16 elements
  return __builtin_shufflevector(lo, hi, 0, 1, 2, 3, 4, 5, 6, 7,
                                          8, 9, 10, 11, 12, 13, 14, 15);
}

__global__ void __launch_bounds__(256)
wmma_gemm_bf16(const __bf16* __restrict__ A, const __bf16* __restrict__ W,
               float* __restrict__ C, const float* __restrict__ bias,
               float bias_scale, int act, int M, int N, int K)
{
  const int lane = threadIdx.x & 31;
  const int wave = threadIdx.x >> 5;
  const int hl   = lane >> 4;          // half-wave
  const int lidx = lane & 15;
  const int m0 = blockIdx.x * 64 + (wave >> 2) * 32;
  const int n0 = blockIdx.y * 128 + (wave & 3) * 32;
  if (m0 >= M || n0 >= N) return;      // wave-uniform guard (EXEC stays all-1)

  const int akoff = hl * 8;            // A: upper half starts at K=8
  const int bkoff = hl * 8;            // B: upper half starts at K=8
  const __bf16* pa0 = A + (size_t)(m0 + lidx) * K + akoff;
  const __bf16* pa1 = A + (size_t)(m0 + 16 + lidx) * K + akoff;
  const __bf16* pb0 = W + (size_t)(n0 + lidx) * K + bkoff;
  const __bf16* pb1 = W + (size_t)(n0 + 16 + lidx) * K + bkoff;

  v8f acc00 = {0,0,0,0,0,0,0,0}, acc01 = {0,0,0,0,0,0,0,0};
  v8f acc10 = {0,0,0,0,0,0,0,0}, acc11 = {0,0,0,0,0,0,0,0};

  for (int k = 0; k < K; k += 32) {
    v16bf a0 = load_afrag(pa0 + k);
    v16bf a1 = load_afrag(pa1 + k);
    v16bf b0 = load_afrag(pb0 + k);
    v16bf b1 = load_afrag(pb1 + k);
    acc00 = __builtin_amdgcn_wmma_f32_16x16x32_bf16(false, a0, false, b0, (short)0, acc00, false, false);
    acc01 = __builtin_amdgcn_wmma_f32_16x16x32_bf16(false, a0, false, b1, (short)0, acc01, false, false);
    acc10 = __builtin_amdgcn_wmma_f32_16x16x32_bf16(false, a1, false, b0, (short)0, acc10, false, false);
    acc11 = __builtin_amdgcn_wmma_f32_16x16x32_bf16(false, a1, false, b1, (short)0, acc11, false, false);
  }

  auto store_tile = [&](const v8f& acc, int mb, int nb) {
    const int col = nb + lidx;
#pragma unroll
    for (int r = 0; r < 8; ++r) {
      float v = acc[r];                         // lane row = hl*8 + r (ISA C layout)
      if (bias) v = fmaf(bias_scale, bias[col], v);
      if (act == 1) v = softplusf_(v);
      C[(size_t)(mb + hl * 8 + r) * N + col] = v;
    }
  };
  store_tile(acc00, m0,      n0);
  store_tile(acc01, m0,      n0 + 16);
  store_tile(acc10, m0 + 16, n0);
  store_tile(acc11, m0 + 16, n0 + 16);
}

// ---------------------------------------------------------------------------
// Small helpers / element-wise kernels
// ---------------------------------------------------------------------------
__global__ void zero_f32(float* __restrict__ p, int n)
{ int i = blockIdx.x * 256 + threadIdx.x; if (i < n) p[i] = 0.0f; }

__global__ void f32_to_bf16(const float* __restrict__ in, __bf16* __restrict__ out, int n)
{ int i = blockIdx.x * 256 + threadIdx.x; if (i < n) out[i] = (__bf16)in[i]; }

// Wsmall (96 x 2048): rows 0-63 = xproj[0:64] (dt_low), 64-79 = xproj[80:96] (C),
// 80-89 = Wc, 90-95 = zero pad.
__global__ void build_wsmall(const float* __restrict__ xproj, const float* __restrict__ Wc,
                             __bf16* __restrict__ Wsmall)
{
  int i = blockIdx.x * 256 + threadIdx.x;
  if (i >= 96 * 2048) return;
  int r = i >> 11, c = i & 2047;
  float v;
  if (r < 64)      v = xproj[r * 2048 + c];
  else if (r < 80) v = xproj[(r + 16) * 2048 + c];
  else if (r < 90) v = Wc[(r - 80) * 2048 + c];
  else             v = 0.0f;
  Wsmall[i] = (__bf16)v;
}

// causal depthwise conv (D_CONV=4) + SiLU; emits fp32 and bf16 copies
__global__ void conv_silu(const float* __restrict__ xz, const float* __restrict__ conv_w,
                          const float* __restrict__ conv_b, float* __restrict__ xconv,
                          __bf16* __restrict__ xconv_bf)
{
  int idx = blockIdx.x * 256 + threadIdx.x;      // BL*2048
  if (idx >= BL * 2048) return;
  int d = idx & 2047;
  int bl = idx >> 11;
  int b = bl >> 9, l = bl & 511;
  float acc = conv_b[d];
#pragma unroll
  for (int k = 0; k < 4; ++k) {
    int lk = l - 3 + k;
    if (lk >= 0) acc += xz[((size_t)(b * 512 + lk)) * 4096 + d] * conv_w[d * 4 + k];
  }
  float v = acc * sigmoidf_(acc);
  xconv[idx] = v;
  xconv_bf[idx] = (__bf16)v;
}

// ---------------------------------------------------------------------------
// GRU recurrence (fp32): per-step GEMV gh = h_{t-1} @ Whh^T + bhh, then gates.
// ---------------------------------------------------------------------------
__global__ void __launch_bounds__(256)
gru_gemv(const float* __restrict__ whh, const float* __restrict__ bhh,
         const float* __restrict__ hgru, const float* __restrict__ h0,
         float* __restrict__ gh, int t)
{
  int gid = blockIdx.x * 32 + (threadIdx.x >> 3);   // 0..12287  (2 batches x 6144)
  int l8  = threadIdx.x & 7;
  int b = (gid >= 6144) ? 1 : 0;
  int j = gid - b * 6144;
  const float* hp = (t == 0) ? (h0 + b * 2048)
                             : (hgru + ((size_t)(b * 512 + t - 1)) * 2048);
  const float4* wp = (const float4*)(whh + (size_t)j * 2048);
  const float4* hv = (const float4*)hp;
  float s = 0.0f;
#pragma unroll 4
  for (int i = l8; i < 512; i += 8) {
    float4 w = wp[i]; float4 h = hv[i];
    s += w.x * h.x + w.y * h.y + w.z * h.z + w.w * h.w;
  }
  s += __shfl_xor(s, 1); s += __shfl_xor(s, 2); s += __shfl_xor(s, 4);
  if (l8 == 0) gh[gid] = s + bhh[j];
}

__global__ void gru_gate(const float* __restrict__ gi, const float* __restrict__ gh,
                         float* __restrict__ hgru, const float* __restrict__ h0, int t)
{
  int idx = blockIdx.x * 256 + threadIdx.x;          // 0..4095 (2 x 2048)
  if (idx >= 2 * 2048) return;
  int b = idx >> 11, j = idx & 2047;
  const float* gib = gi + ((size_t)(b * 512 + t)) * 6144;
  const float* ghb = gh + b * 6144;
  float r  = sigmoidf_(gib[j] + ghb[j]);
  float u  = sigmoidf_(gib[2048 + j] + ghb[2048 + j]);
  float nn = tanhf(gib[4096 + j] + r * ghb[4096 + j]);
  float hp = (t == 0) ? h0[idx] : hgru[((size_t)(b * 512 + t - 1)) * 2048 + j];
  hgru[((size_t)(b * 512 + t)) * 2048 + j] = (1.0f - u) * nn + u * hp;
}

// scaling = clip(softplus(clip(h_gru @ WT^T + b, -10, 10)), 0.1, 10); wave per (b,l)
__global__ void __launch_bounds__(256)
scaling_kernel(const float* __restrict__ hgru, const float* __restrict__ WT_w,
               const float* __restrict__ WT_b, float* __restrict__ scaling)
{
  int wid  = blockIdx.x * 8 + (threadIdx.x >> 5);    // 0..1023
  int lane = threadIdx.x & 31;
  const float* h = hgru + (size_t)wid * 2048;
  float s = 0.0f;
  for (int i = lane; i < 2048; i += 32) s += h[i] * WT_w[i];
  s += __shfl_xor(s, 1); s += __shfl_xor(s, 2); s += __shfl_xor(s, 4);
  s += __shfl_xor(s, 8); s += __shfl_xor(s, 16);
  if (lane == 0) {
    float raw = fminf(fmaxf(s + WT_b[0], -10.0f), 10.0f);
    scaling[wid] = fminf(fmaxf(softplusf_(raw), 0.1f), 10.0f);
  }
}

// intervals = diff(delta*scaling) along L; temporal half of B_dual; out tail
__global__ void intervals_kernel(const float* __restrict__ delta,
                                 const float* __restrict__ scaling,
                                 const float* __restrict__ Wtau,
                                 const float* __restrict__ B0,
                                 float* __restrict__ intervals,
                                 float* __restrict__ bdual,
                                 float* __restrict__ out_tail)
{
  int idx = blockIdx.x * 256 + threadIdx.x;          // 0..1023
  if (idx >= BL) return;
  int b = idx >> 9, l = idx & 511;
  float stg  = delta[idx] * scaling[idx];
  float prev = (l == 0) ? 0.0f : delta[idx - 1] * scaling[idx - 1];
  intervals[idx] = stg - prev;
#pragma unroll
  for (int i = 0; i < 6; ++i)
    bdual[idx * 16 + 10 + i] = sigmoidf_(stg * Wtau[i]) * B0[10 + i];
  if (l == 511) out_tail[b] = scaling[idx];
}

// content half of B_dual from smallout cols 80..89
__global__ void bdual_content(const float* __restrict__ smallout,
                              const float* __restrict__ B0, float* __restrict__ bdual)
{
  int idx = blockIdx.x * 256 + threadIdx.x;          // BL*10
  if (idx >= BL * 10) return;
  int bl = idx / 10, i = idx - bl * 10;
  bdual[bl * 16 + i] = sigmoidf_(smallout[bl * 96 + 80 + i]) * B0[i];
}

__global__ void dtlow_extract(const float* __restrict__ smallout, __bf16* __restrict__ dtlow)
{
  int idx = blockIdx.x * 256 + threadIdx.x;          // BL*64
  if (idx >= BL * 64) return;
  int bl = idx >> 6, r = idx & 63;
  dtlow[idx] = (__bf16)smallout[bl * 96 + r];
}

// ---------------------------------------------------------------------------
// Fused selective-scan: 16 lanes = 16 states per (b,d) channel; emits y in bf16
// ---------------------------------------------------------------------------
__global__ void __launch_bounds__(256)
scan_kernel(const float* __restrict__ intervals, const float* __restrict__ basedt,
            const float* __restrict__ bdual, const float* __restrict__ smallout,
            const float* __restrict__ xconv, const float* __restrict__ xz,
            const float* __restrict__ A_log, const float* __restrict__ D_param,
            __bf16* __restrict__ y_bf)
{
  int g = blockIdx.x * 16 + (threadIdx.x >> 4);      // channel 0..4095
  int n = threadIdx.x & 15;                          // state index
  int b = g >> 11;
  int d = g & 2047;
  float Av = -expf(A_log[d * 16 + n]);
  float Dp = D_param[d];
  float h = 0.0f;
  for (int l = 0; l < LL; ++l) {
    int bl = b * LL + l;
    float dts = intervals[bl] * basedt[(size_t)bl * 2048 + d];
    dts = fminf(fmaxf(dts, 1e-6f), 10.0f);
    float e = fminf(fmaxf(dts * Av, -20.0f), 20.0f);
    float xc = xconv[(size_t)bl * 2048 + d];
    h = expf(e) * h + dts * bdual[bl * 16 + n] * xc;
    float yv = h * smallout[bl * 96 + 64 + n];       // C = cols 64..79
    yv += __shfl_xor(yv, 1); yv += __shfl_xor(yv, 2);
    yv += __shfl_xor(yv, 4); yv += __shfl_xor(yv, 8);
    if (n == 0) {
      float z = xz[(size_t)bl * 4096 + 2048 + d];
      float yo = (yv + xc * Dp) * (z * sigmoidf_(z));
      y_bf[(size_t)bl * 2048 + d] = (__bf16)yo;
    }
  }
}

// ---------------------------------------------------------------------------
extern "C" void kernel_launch(void* const* d_in, const int* in_sizes, int n_in,
                              void* d_out, int out_size, void* d_ws, size_t ws_size,
                              hipStream_t stream)
{
  const float* x         = (const float*)d_in[0];
  const float* delta     = (const float*)d_in[1];
  const float* in_proj_w = (const float*)d_in[2];
  const float* conv_w    = (const float*)d_in[3];
  const float* conv_b    = (const float*)d_in[4];
  const float* gru_wih   = (const float*)d_in[5];
  const float* gru_whh   = (const float*)d_in[6];
  const float* gru_bih   = (const float*)d_in[7];
  const float* gru_bhh   = (const float*)d_in[8];
  const float* WT_w      = (const float*)d_in[9];
  const float* WT_b      = (const float*)d_in[10];
  const float* Wc_w      = (const float*)d_in[11];
  const float* Wtau_w    = (const float*)d_in[12];
  const float* B0        = (const float*)d_in[13];
  const float* xproj_w   = (const float*)d_in[14];
  const float* dtproj_w  = (const float*)d_in[15];
  const float* dtproj_b  = (const float*)d_in[16];
  const float* A_log     = (const float*)d_in[17];
  const float* D_param   = (const float*)d_in[18];
  const float* outproj_w = (const float*)d_in[19];
  float* out = (float*)d_out;

  char* ws = (char*)d_ws; size_t off = 0;
  auto take = [&](size_t bytes) -> void* {
    void* p = ws + off; off += (bytes + 255) & ~(size_t)255; return p;
  };
  float*  xz        = (float*)take((size_t)BL * 4096 * 4);
  float*  xconv     = (float*)take((size_t)BL * 2048 * 4);
  float*  gi        = (float*)take((size_t)BL * 6144 * 4);
  float*  hgru      = (float*)take((size_t)BL * 2048 * 4);
  float*  gh        = (float*)take((size_t)2 * 6144 * 4);
  float*  h0        = (float*)take((size_t)2 * 2048 * 4);
  float*  scaling   = (float*)take((size_t)BL * 4);
  float*  intervals = (float*)take((size_t)BL * 4);
  float*  smallout  = (float*)take((size_t)BL * 96 * 4);
  float*  basedt    = (float*)take((size_t)BL * 2048 * 4);
  float*  bdual     = (float*)take((size_t)BL * 16 * 4);
  __bf16* x_bf      = (__bf16*)take((size_t)BL * 1024 * 2);
  __bf16* xconv_bf  = (__bf16*)take((size_t)BL * 2048 * 2);
  __bf16* dtlow_bf  = (__bf16*)take((size_t)BL * 64 * 2);
  __bf16* y_bf      = (__bf16*)take((size_t)BL * 2048 * 2);
  __bf16* Wxz_bf    = (__bf16*)take((size_t)4096 * 1024 * 2);
  __bf16* Wih_bf    = (__bf16*)take((size_t)6144 * 2048 * 2);
  __bf16* Wsm_bf    = (__bf16*)take((size_t)96 * 2048 * 2);
  __bf16* Wdt_bf    = (__bf16*)take((size_t)2048 * 64 * 2);
  __bf16* Wout_bf   = (__bf16*)take((size_t)1024 * 2048 * 2);

  auto cdiv = [](int a, int b) { return (a + b - 1) / b; };

  // ---- conversions / init ----
  zero_f32   <<<cdiv(4096, 256), 256, 0, stream>>>(h0, 4096);
  f32_to_bf16<<<cdiv(BL * 1024, 256), 256, 0, stream>>>(x, x_bf, BL * 1024);
  f32_to_bf16<<<cdiv(4096 * 1024, 256), 256, 0, stream>>>(in_proj_w, Wxz_bf, 4096 * 1024);
  f32_to_bf16<<<cdiv(6144 * 2048, 256), 256, 0, stream>>>(gru_wih, Wih_bf, 6144 * 2048);
  f32_to_bf16<<<cdiv(2048 * 64, 256), 256, 0, stream>>>(dtproj_w, Wdt_bf, 2048 * 64);
  f32_to_bf16<<<cdiv(1024 * 2048, 256), 256, 0, stream>>>(outproj_w, Wout_bf, 1024 * 2048);
  build_wsmall<<<cdiv(96 * 2048, 256), 256, 0, stream>>>(xproj_w, Wc_w, Wsm_bf);

  // ---- parallel GEMMs (WMMA bf16) ----
  wmma_gemm_bf16<<<dim3(BL / 64, 4096 / 128), 256, 0, stream>>>(
      x_bf, Wxz_bf, xz, nullptr, 0.0f, 0, BL, 4096, 1024);
  conv_silu<<<cdiv(BL * 2048, 256), 256, 0, stream>>>(xz, conv_w, conv_b, xconv, xconv_bf);
  wmma_gemm_bf16<<<dim3(BL / 64, 6144 / 128), 256, 0, stream>>>(
      xconv_bf, Wih_bf, gi, gru_bih, 1.0f, 0, BL, 6144, 2048);
  wmma_gemm_bf16<<<dim3(BL / 64, 1), 256, 0, stream>>>(
      xconv_bf, Wsm_bf, smallout, nullptr, 0.0f, 0, BL, 96, 2048);
  dtlow_extract<<<cdiv(BL * 64, 256), 256, 0, stream>>>(smallout, dtlow_bf);
  bdual_content<<<cdiv(BL * 10, 256), 256, 0, stream>>>(smallout, B0, bdual);
  wmma_gemm_bf16<<<dim3(BL / 64, 2048 / 128), 256, 0, stream>>>(
      dtlow_bf, Wdt_bf, basedt, dtproj_b, 2.0f, 1, BL, 2048, 64);

  // ---- sequential GRU (fp32) ----
  for (int t = 0; t < LL; ++t) {
    gru_gemv<<<384, 256, 0, stream>>>(gru_whh, gru_bhh, hgru, h0, gh, t);
    gru_gate<<<16, 256, 0, stream>>>(gi, gh, hgru, h0, t);
  }
  scaling_kernel <<<128, 256, 0, stream>>>(hgru, WT_w, WT_b, scaling);
  intervals_kernel<<<cdiv(BL, 256), 256, 0, stream>>>(
      delta, scaling, Wtau_w, B0, intervals, bdual, out + (size_t)BL * 1024);

  // ---- fused selective scan + epilogue ----
  scan_kernel<<<256, 256, 0, stream>>>(intervals, basedt, bdual, smallout,
                                       xconv, xz, A_log, D_param, y_bf);

  // ---- out projection (WMMA bf16) ----
  wmma_gemm_bf16<<<dim3(BL / 64, 1024 / 128), 256, 0, stream>>>(
      y_bf, Wout_bf, out, nullptr, 0.0f, 0, BL, 1024, 2048);
}
